// ComplexLSTMLayer_83184926589417
// MI455X (gfx1250) — matrix operand order repaired and checked by reading, not compiled, not run on hardware
//
#include <hip/hip_runtime.h>
#include <hip/hip_bf16.h>

// ---------------------------------------------------------------------------
// ComplexLSTM for gfx1250 (MI455X): bf16 WMMA GEMMs + persistent scan kernel.
// B=32, T=1024, I=H=512.  Combined K = 1024 ([real|imag]), combined N = 2560
// (gate columns [i | f | o | c_real | c_imag], 512 each).
// ---------------------------------------------------------------------------

#define BB   32
#define TT   1024
#define HH   512
#define KC   1024          // combined K (real|imag)
#define NC   2560          // combined N (5 gates)
#define MM   (BB * TT)     // 32768
#define KT32 (KC / 32)     // 32 k-tiles of 32
#define NT16 (NC / 16)     // 160 n-tiles of 16
#define NWG_RECUR 8

typedef __attribute__((ext_vector_type(16))) __bf16 v16bf;
typedef __attribute__((ext_vector_type(8)))  float  v8f;

__device__ __forceinline__ v8f wmma_bf16(v16bf a, v16bf b, v8f c) {
    // D = A(16x32) * B(32x16) + C, fp32 accumulate
    return __builtin_amdgcn_wmma_f32_16x16x32_bf16(
        /*neg_a=*/false, a, /*neg_b=*/false, b,
        /*c_mod=*/(short)0, c, /*reuse_a=*/false, /*reuse_b=*/false);
}

// ---------------------------------------------------------------------------
// 1) z (B,T,I,2) fp32  ->  Zc (M x 1024) bf16 row-major, [zr | zi]
// ---------------------------------------------------------------------------
__global__ __launch_bounds__(256) void cvt_z_kernel(const float* __restrict__ z,
                                                    __bf16* __restrict__ Zc) {
    size_t i = (size_t)blockIdx.x * 256 + threadIdx.x;   // over M*KC
    size_t m = i >> 10;
    int k    = (int)(i & 1023);
    int comp = k >> 9;           // 0 = real, 1 = imag
    int kk   = k & 511;
    Zc[i] = (__bf16)z[(m * 512 + kk) * 2 + comp];
}

// ---------------------------------------------------------------------------
// 2) Build combined weight matrices in WMMA B-fragment tiling.
//    Fragment(nt,kt): 32 lanes x 16 bf16, element e of lane l is
//    W[kt*32 + (l<16?0:8) + (e<8?e:e+8)][nt*16 + (l&15)].
//    Flat offset = (nt*KT32 + kt)*512 + l*16 + e  == global thread index.
// ---------------------------------------------------------------------------
__global__ __launch_bounds__(256) void build_u_kernel(
    const float* __restrict__ Wi, const float* __restrict__ Wf,
    const float* __restrict__ Wo, const float* __restrict__ Wc,
    const float* __restrict__ Hi, const float* __restrict__ Hf,
    const float* __restrict__ Ho, const float* __restrict__ Hc,
    __bf16* __restrict__ U, __bf16* __restrict__ Uh) {
    size_t idx = (size_t)blockIdx.x * 256 + threadIdx.x;  // over NT16*KT32*512
    int within = (int)(idx & 511);
    size_t frag = idx >> 9;
    int l = within >> 4, e = within & 15;
    int nt = (int)(frag / KT32), kt = (int)(frag % KT32);
    int n = nt * 16 + (l & 15);
    int k = kt * 32 + ((l < 16) ? 0 : 8) + ((e < 8) ? e : e + 8);

    int g = n >> 9, u = n & 511, kk = k & 511, top = k >> 9;

    const float *Wz, *Wr;
    if (g == 0)      { Wz = Wi; Wr = Hi; }
    else if (g == 1) { Wz = Wf; Wr = Hf; }
    else if (g == 2) { Wz = Wo; Wr = Ho; }
    else             { Wz = Wc; Wr = Hc; }   // g==3 (cr) and g==4 (ci)

    size_t widx = ((size_t)kk * 512 + u) * 2;
    float zr_ = Wz[widx], zi_ = Wz[widx + 1];
    float hr_ = Wr[widx], hi_ = Wr[widx + 1];

    float vz, vh;
    if (g <= 3) {            // real-part projection: [W_r ; -W_i]
        vz = (top == 0) ? zr_ : -zi_;
        vh = (top == 0) ? hr_ : -hi_;
    } else {                 // imag cell column: [W_i ; W_r]
        vz = (top == 0) ? zi_ : zr_;
        vh = (top == 0) ? hi_ : hr_;
    }
    U[idx]  = (__bf16)vz;
    Uh[idx] = (__bf16)vh;
}

// ---------------------------------------------------------------------------
// 3) Input projection GEMM: X(M x 2560) = Zc(M x 1024) @ U, fp32 out.
//    Block 128x128, 8 waves, wave tile 32x64. A staged via LDS in A-fragment
//    layout; B streamed from tiled global (32B/lane contiguous).
// ---------------------------------------------------------------------------
__global__ __launch_bounds__(256) void gemm_in_kernel(const __bf16* __restrict__ Zc,
                                                      const __bf16* __restrict__ Ut,
                                                      float* __restrict__ X) {
    __shared__ __align__(32) __bf16 sA[4 * 32 * 16];   // 4 A-fragments (64x32 tile)

    const int tid  = threadIdx.x;
    const int lane = tid & 31;
    const int w    = tid >> 5;
    const int wm   = w & 3;      // 4 wave rows  (M)
    const int wn   = w >> 2;     // 2 wave cols  (N)

    const size_t Mbase  = (size_t)blockIdx.y * 128;
    const int    Ntbase = blockIdx.x * 8;              // 8 n-tiles per block

    // cooperative A-tile staging: 256 threads, one uint4 (8 bf16) each
    const int f     = tid >> 6;          // fragment (16-row group)
    const int lslot = (tid >> 1) & 31;   // lane slot inside fragment
    const int half  = tid & 1;           // element half (e<8 / e>=8)
    const __bf16* gsrc = Zc + (Mbase + f * 16 + (lslot & 15)) * (size_t)KC
                            + half * 16 + ((lslot < 16) ? 0 : 8);
    __bf16* ldst = &sA[(f * 32 + lslot) * 16 + half * 8];

    v8f acc[2][4] = {};

    for (int kc = 0; kc < KT32; ++kc) {
        __syncthreads();
        *(uint4*)ldst = *(const uint4*)(gsrc + kc * 32);
        __syncthreads();

        v16bf a0 = *(const v16bf*)&sA[((2 * wm + 0) * 32 + lane) * 16];
        v16bf a1 = *(const v16bf*)&sA[((2 * wm + 1) * 32 + lane) * 16];
#pragma unroll
        for (int j = 0; j < 4; ++j) {
            const int nt = Ntbase + wn * 4 + j;
            const __bf16* bp = Ut + ((size_t)nt * KT32 + kc) * 512 + lane * 16;
            v16bf bv = *(const v16bf*)bp;
            if (kc + 1 < KT32) __builtin_prefetch(bp + 512, 0, 0);  // next K fragment
            acc[0][j] = wmma_bf16(a0, bv, acc[0][j]);
            acc[1][j] = wmma_bf16(a1, bv, acc[1][j]);
        }
    }

#pragma unroll
    for (int mt = 0; mt < 2; ++mt) {
#pragma unroll
        for (int j = 0; j < 4; ++j) {
            size_t m0 = Mbase + (2 * wm + mt) * 16 + ((lane < 16) ? 0 : 8);
            int n = (Ntbase + wn * 4 + j) * 16 + (lane & 15);
#pragma unroll
            for (int v = 0; v < 8; ++v)
                X[(m0 + v) * (size_t)NC + n] = acc[mt][j][v];
        }
    }
}

// ---------------------------------------------------------------------------
// 4) Persistent recurrence: 8 WGs x 256 thr = 64 waves.
//    Wave (ub, mtile): 16 hidden units x 16 batch rows, 5 gate C-fragments.
//    Per step: 32 kc x 5 WMMA, lane-local pointwise, grid barrier.
// ---------------------------------------------------------------------------
__global__ __launch_bounds__(256) void recur_kernel(
    const __bf16* __restrict__ Uh, const float* __restrict__ X,
    const float* __restrict__ b_i, const float* __restrict__ b_f,
    const float* __restrict__ b_o, const float* __restrict__ b_c,
    __bf16* __restrict__ Hc0, __bf16* __restrict__ Hc1,
    float* __restrict__ cst, unsigned* __restrict__ cnt,
    float* __restrict__ out) {
    const int tid  = threadIdx.x;
    const int lane = tid & 31;
    const int w    = blockIdx.x * 8 + (tid >> 5);   // 0..63
    const int ub    = w >> 1;                       // unit block 0..31
    const int mtile = w & 1;                        // batch half

    const int u     = ub * 16 + (lane & 15);
    const int arow  = mtile * 16 + (lane & 15);     // A-fragment row (batch)
    const int mbase = mtile * 16 + ((lane < 16) ? 0 : 8);  // D rows base

    const float bi  = b_i[u], bf_ = b_f[u], bo = b_o[u];
    const float bcr = b_c[2 * u], bci = b_c[2 * u + 1];

    const size_t HSEQ = (size_t)BB * TT * HH * 2;   // hidden_seq elements

    for (int t = 0; t < TT; ++t) {
        const __bf16* Hin  = (t & 1) ? Hc1 : Hc0;
        __bf16*       Hout = (t & 1) ? Hc0 : Hc1;

        v8f acc[5] = {};
        const __bf16* ap = Hin + (size_t)arow * KC + ((lane < 16) ? 0 : 8);
#pragma unroll 4
        for (int kc = 0; kc < KT32; ++kc) {
            union { v16bf v; uint4 q[2]; } au;
            au.q[0] = *(const uint4*)(ap + kc * 32);        // K = base .. base+7
            au.q[1] = *(const uint4*)(ap + kc * 32 + 16);   // K = base+16 .. +23
#pragma unroll
            for (int g = 0; g < 5; ++g) {
                const __bf16* bp = Uh + ((size_t)(g * 32 + ub) * KT32 + kc) * 512
                                      + lane * 16;
                v16bf bv = *(const v16bf*)bp;
                acc[g] = wmma_bf16(au.v, bv, acc[g]);
            }
        }

        // lane-local pointwise: all 5 gates of unit u sit in the same slot
#pragma unroll
        for (int v = 0; v < 8; ++v) {
            const int b = mbase + v;                       // batch 0..31
            const size_t xrow = ((size_t)b * TT + t) * NC;
            float gi = acc[0][v] + X[xrow + u]            + bi;
            float gf = acc[1][v] + X[xrow + 512 + u]      + bf_;
            float go = acc[2][v] + X[xrow + 1024 + u]     + bo;
            float cr = acc[3][v] + X[xrow + 1536 + u]     + bcr;
            float ci = acc[4][v] + X[xrow + 2048 + u]     + bci;

            float it = 1.0f / (1.0f + expf(-gi));
            float ft = 1.0f / (1.0f + expf(-gf));
            float ot = 1.0f / (1.0f + expf(-go));

            float mag = fmaxf(sqrtf(cr * cr + ci * ci + 1e-14f), 1e-8f);
            float th  = tanhf(mag) / mag;
            float ctr = th * cr, cti = th * ci;

            const size_t cidx = ((size_t)b * HH + u) * 2;
            float c_r = ft * cst[cidx]     + it * ctr;
            float c_i = ft * cst[cidx + 1] + it * cti;
            cst[cidx] = c_r; cst[cidx + 1] = c_i;

            float mag2 = fmaxf(sqrtf(c_r * c_r + c_i * c_i + 1e-14f), 1e-8f);
            float th2  = tanhf(mag2) / mag2;
            float hr = ot * th2 * c_r;
            float hi = ot * th2 * c_i;

            const size_t hidx = (((size_t)b * TT + t) * HH + u) * 2;
            out[hidx] = hr; out[hidx + 1] = hi;

            Hout[(size_t)b * KC + u]       = (__bf16)hr;
            Hout[(size_t)b * KC + 512 + u] = (__bf16)hi;

            if (t == TT - 1) {
                const size_t s = ((size_t)b * HH + u) * 2;
                out[HSEQ + s]     = hr;  out[HSEQ + s + 1] = hi;
                out[HSEQ + (size_t)BB * HH * 2 + s]     = c_r;
                out[HSEQ + (size_t)BB * HH * 2 + s + 1] = c_i;
            }
        }

        // device-scope monotonic barrier (8 workgroups)
        __threadfence();
        __syncthreads();
        if (tid == 0) {
            __hip_atomic_fetch_add(cnt, 1u, __ATOMIC_ACQ_REL, __HIP_MEMORY_SCOPE_AGENT);
            const unsigned target = (unsigned)NWG_RECUR * (unsigned)(t + 1);
            while (__hip_atomic_load(cnt, __ATOMIC_ACQUIRE, __HIP_MEMORY_SCOPE_AGENT) < target)
                __builtin_amdgcn_s_sleep(2);
        }
        __syncthreads();
    }
}

// ---------------------------------------------------------------------------
// launch
// ---------------------------------------------------------------------------
extern "C" void kernel_launch(void* const* d_in, const int* in_sizes, int n_in,
                              void* d_out, int out_size, void* d_ws, size_t ws_size,
                              hipStream_t stream) {
    (void)in_sizes; (void)n_in; (void)out_size; (void)ws_size;

    const float* z    = (const float*)d_in[0];
    const float* W_zi = (const float*)d_in[1];
    const float* W_hi = (const float*)d_in[2];
    const float* b_i  = (const float*)d_in[3];
    const float* W_zf = (const float*)d_in[4];
    const float* W_hf = (const float*)d_in[5];
    const float* b_f  = (const float*)d_in[6];
    const float* W_zc = (const float*)d_in[7];
    const float* W_hc = (const float*)d_in[8];
    const float* b_c  = (const float*)d_in[9];
    const float* W_zo = (const float*)d_in[10];
    const float* W_ho = (const float*)d_in[11];
    const float* b_o  = (const float*)d_in[12];
    float* out = (float*)d_out;

    // workspace layout (bytes, 256-aligned sections)
    char* ws = (char*)d_ws;
    const size_t ZC_B  = (size_t)MM * KC * 2;        // 67,108,864
    const size_t U_B   = (size_t)KC * NC * 2;        //  5,242,880
    const size_t X_B   = (size_t)MM * NC * 4;        // 335,544,320
    const size_t HC_B  = (size_t)BB * KC * 2;        //     65,536
    const size_t CST_B = (size_t)BB * HH * 2 * 4;    //    131,072

    __bf16* Zc  = (__bf16*)(ws);
    __bf16* U   = (__bf16*)(ws + ZC_B);
    __bf16* Uh  = (__bf16*)(ws + ZC_B + U_B);
    float*  X   = (float*) (ws + ZC_B + 2 * U_B);
    char*   st  =          ws + ZC_B + 2 * U_B + X_B;
    __bf16* Hc0 = (__bf16*)(st);
    __bf16* Hc1 = (__bf16*)(st + HC_B);
    float*  cst = (float*) (st + 2 * HC_B);
    unsigned* cnt = (unsigned*)(st + 2 * HC_B + CST_B);

    // zero h0 / c0 / barrier counter (graph-capture-safe async memset)
    hipMemsetAsync(st, 0, 2 * HC_B + CST_B + 128, stream);

    // 1) complexify + bf16 convert z
    cvt_z_kernel<<<(MM * (size_t)KC) / 256, 256, 0, stream>>>(z, Zc);

    // 2) build combined/tiled bf16 weights (input + recurrent)
    build_u_kernel<<<((size_t)NT16 * KT32 * 512) / 256, 256, 0, stream>>>(
        W_zi, W_zf, W_zo, W_zc, W_hi, W_hf, W_ho, W_hc, U, Uh);

    // 3) input projection GEMM (M=32768, K=1024, N=2560)
    dim3 ggrid(NC / 128, MM / 128);   // 20 x 256
    gemm_in_kernel<<<ggrid, 256, 0, stream>>>(Zc, U, X);

    // 4) persistent sequential scan
    recur_kernel<<<NWG_RECUR, 256, 0, stream>>>(Uh, X, b_i, b_f, b_o, b_c,
                                                Hc0, Hc1, cst, cnt, out);
}
